// CapsuleLayer_74569222193578
// MI455X (gfx1250) — compile-verified, matrix operand order checked by reading, not run on hardware
//
#include <hip/hip_runtime.h>

// Capsule routing, fused for MI455X (gfx1250, wave32, WMMA bf16 16x16x32).
// u_hat is never materialized in HBM. Phase B packs FOUR n-values into the
// four 8-wide K-blocks of one v_wmma_f32_16x16x32_bf16 (coupling coefficient
// c is folded into the x operand), so the weighted sum over n runs at full
// K utilization with no exec masking. Phase A (routing logits) needs per-n
// tiles, so it uses the K-padded single-n form.
//
// Workspace layout (24 MiB):
//   [0, 8 MiB)   : spart[256 chunks][32 j][16 d][16 b] fp32 per-chunk partial s
//   [8, 24 MiB)  : Lws[32 j][2048 n][64 b] fp32 routing logits

#define B_ 64
#define N_ 2048
#define I_ 8
#define J_ 32
#define D_ 16

typedef __attribute__((ext_vector_type(16))) __bf16          v16bf;
typedef __attribute__((ext_vector_type(16))) unsigned short  v16us;
typedef __attribute__((ext_vector_type(8)))  float           v8f;

union ABTile { v16bf bf; v16us us; };

static __device__ __forceinline__ unsigned short f2bf(float f) {
  return (unsigned short)(__float_as_uint(f) >> 16);  // bf16 truncate
}

// Single-n A operand: rows = d of one j, K = i (8 real, 24 zero).
static __device__ __forceinline__ ABTile load_w_tile(const float* __restrict__ W,
                                                     int j, int n, int lane) {
  ABTile a;
  #pragma unroll
  for (int k = 0; k < 16; ++k) a.us[k] = 0;
  if (lane < 16) {
    const float* wp = W + (((size_t)j * N_ + n) * D_ + lane) * I_;
    float4 w0 = *(const float4*)wp;
    float4 w1 = *(const float4*)(wp + 4);
    a.us[0] = f2bf(w0.x); a.us[1] = f2bf(w0.y); a.us[2] = f2bf(w0.z); a.us[3] = f2bf(w0.w);
    a.us[4] = f2bf(w1.x); a.us[5] = f2bf(w1.y); a.us[6] = f2bf(w1.z); a.us[7] = f2bf(w1.w);
  }
  return a;
}

// 4-n packed A operand: K-blocks {0-7,8-15,16-23,24-31} carry n_lo (this
// lane-half) in elements 0..7 and n_hi = n_lo+2 in elements 8..15.
// All 32 lanes load: row d = lane&15, lane-half selects n parity.
static __device__ __forceinline__ ABTile load_w_tile4(const float* __restrict__ W,
                                                      int j, int n_lo, int n_hi, int d) {
  const float* wp0 = W + (((size_t)j * N_ + n_lo) * D_ + d) * I_;
  const float* wp1 = W + (((size_t)j * N_ + n_hi) * D_ + d) * I_;
  float4 a0 = *(const float4*)wp0;
  float4 a1 = *(const float4*)(wp0 + 4);
  float4 b0 = *(const float4*)wp1;
  float4 b1 = *(const float4*)(wp1 + 4);
  ABTile a;
  a.us[0]  = f2bf(a0.x); a.us[1]  = f2bf(a0.y); a.us[2]  = f2bf(a0.z); a.us[3]  = f2bf(a0.w);
  a.us[4]  = f2bf(a1.x); a.us[5]  = f2bf(a1.y); a.us[6]  = f2bf(a1.z); a.us[7]  = f2bf(a1.w);
  a.us[8]  = f2bf(b0.x); a.us[9]  = f2bf(b0.y); a.us[10] = f2bf(b0.z); a.us[11] = f2bf(b0.w);
  a.us[12] = f2bf(b1.x); a.us[13] = f2bf(b1.y); a.us[14] = f2bf(b1.z); a.us[15] = f2bf(b1.w);
  return a;
}

// One workgroup = 16 batches (blockIdx.y) x 32 n (blockIdx.x), 4 waves x 8 n.
// MODE 0: iter0 (c uniform 1/32, WMMA-chained over both 4-n groups)
// MODE 1: iter1 (logits = u_hat . v0, stored to Lws), softmax, weighted s
// MODE 2: iter2 (logits = Lws + u_hat . v1), softmax, weighted s
template <int MODE>
__global__ void __launch_bounds__(128)
caps_route(const float* __restrict__ x, const float* __restrict__ W,
           const float* __restrict__ vprev, float* __restrict__ spart,
           float* __restrict__ Lws) {
  extern __shared__ float smem[];
  float* ldsS = smem;          // 4 waves * 8 j * 256 = 8192 floats (32 KiB)
  float* ldsL = smem + 8192;   // 32 n * 32 j * 16 b  = 16384 floats (64 KiB)

  const int tid  = threadIdx.x;
  const int w    = tid >> 5;
  const int lane = tid & 31;
  const int bl   = lane & 15;
  const int hi   = lane >> 4;
  const int b0   = blockIdx.y * 16;
  const int n0   = blockIdx.x * 32;
  const int nw   = n0 + w * 8;

  // x for this wave's 8 n, in the 4-n packed layout (all lanes useful):
  // group G: n_lo = nw + 4G + hi, n_hi = n_lo + 2; col b = bl.
  float xf[2][16];
  #pragma unroll
  for (int G = 0; G < 2; ++G) {
    const int n_lo = nw + 4 * G + hi;
    const float* xp0 = x + ((size_t)(b0 + bl) * N_ + n_lo) * I_;
    const float* xp1 = xp0 + 2 * I_;  // n_hi = n_lo + 2
    float4 a0 = *(const float4*)xp0;
    float4 a1 = *(const float4*)(xp0 + 4);
    float4 c0 = *(const float4*)xp1;
    float4 c1 = *(const float4*)(xp1 + 4);
    xf[G][0]  = a0.x; xf[G][1]  = a0.y; xf[G][2]  = a0.z; xf[G][3]  = a0.w;
    xf[G][4]  = a1.x; xf[G][5]  = a1.y; xf[G][6]  = a1.z; xf[G][7]  = a1.w;
    xf[G][8]  = c0.x; xf[G][9]  = c0.y; xf[G][10] = c0.z; xf[G][11] = c0.w;
    xf[G][12] = c1.x; xf[G][13] = c1.y; xf[G][14] = c1.z; xf[G][15] = c1.w;
  }

  v8f vzero;
  #pragma unroll
  for (int r = 0; r < 8; ++r) vzero[r] = 0.0f;

  // Single-n x tiles for phase A (K-padded, lanes 0-15 only). Only needed
  // when routing logits are computed (MODE >= 1).
  ABTile bx[8];
  if (MODE >= 1) {
    #pragma unroll
    for (int t = 0; t < 8; ++t) {
      #pragma unroll
      for (int k = 0; k < 16; ++k) bx[t].us[k] = 0;
    }
    if (lane < 16) {
      #pragma unroll
      for (int t = 0; t < 8; ++t) {
        const float* xp = x + ((size_t)(b0 + bl) * N_ + (nw + t)) * I_;
        float4 x0 = *(const float4*)xp;
        float4 x1 = *(const float4*)(xp + 4);
        bx[t].us[0] = f2bf(x0.x); bx[t].us[1] = f2bf(x0.y);
        bx[t].us[2] = f2bf(x0.z); bx[t].us[3] = f2bf(x0.w);
        bx[t].us[4] = f2bf(x1.x); bx[t].us[5] = f2bf(x1.y);
        bx[t].us[6] = f2bf(x1.z); bx[t].us[7] = f2bf(x1.w);
      }
    }

    // ---- Phase A: logits a[b,j,n] = sum_d u_hat * v_prev, then softmax_j ----
    for (int j = 0; j < J_; ++j) {
      float vr[8];  // v_prev replicated in C-tile layout: d = r + 8*hi, b = bl
      #pragma unroll
      for (int r = 0; r < 8; ++r)
        vr[r] = vprev[(size_t)(b0 + bl) * (J_ * D_) + j * D_ + r + 8 * hi];
      #pragma unroll
      for (int t = 0; t < 8; ++t) {
        const int n = nw + t;
        ABTile wa = load_w_tile(W, j, n, lane);
        v8f tile = __builtin_amdgcn_wmma_f32_16x16x32_bf16(
            false, wa.bf, false, bx[t].bf, (short)0, vzero, false, false);
        float p = 0.0f;
        #pragma unroll
        for (int r = 0; r < 8; ++r) p += tile[r] * vr[r];
        p += __shfl_xor(p, 16, 32);  // combine d halves -> a[b] in both halves
        if (lane < 16) {
          const size_t loff = ((size_t)j * N_ + n) * B_ + b0 + lane;
          float Lv = p;
          if (MODE == 2) Lv += Lws[loff];   // b += u_hat . v
          else           Lws[loff] = Lv;    // first logit write (iter1)
          ldsL[((n - n0) * J_ + j) * 16 + lane] = Lv;
        }
      }
    }
    __syncthreads();
    // Softmax over j for each (b, n) of this chunk (c written back into ldsL).
    for (int t = 0; t < 4; ++t) {
      const int n_loc = w * 8 + t * 2 + hi;
      float* Lp = &ldsL[(n_loc * J_) * 16 + bl];
      float m = -3.0e38f;
      for (int j = 0; j < J_; ++j) m = fmaxf(m, Lp[j * 16]);
      float ssum = 0.0f;
      for (int j = 0; j < J_; ++j) {
        float e = __expf(Lp[j * 16] - m);
        Lp[j * 16] = e;
        ssum += e;
      }
      const float inv = 1.0f / ssum;
      for (int j = 0; j < J_; ++j) Lp[j * 16] *= inv;
    }
    __syncthreads();
  }

  // Pre-packed (unscaled) bf16 x for MODE 0 — reused across all j.
  ABTile bx4[2];
  if (MODE == 0) {
    #pragma unroll
    for (int G = 0; G < 2; ++G)
      #pragma unroll
      for (int k = 0; k < 16; ++k) bx4[G].us[k] = f2bf(xf[G][k]);
  }

  // ---- Phase B: s[b,j,d] partials = sum_n c * u_hat, 4 n per WMMA ----
  const int p_idx = blockIdx.y * 64 + blockIdx.x;
  for (int g = 0; g < 4; ++g) {          // j groups of 8 so ldsS stays 32 KiB
    for (int jg = 0; jg < 8; ++jg) {
      const int j = g * 8 + jg;
      v8f acc = vzero;
      #pragma unroll
      for (int G = 0; G < 2; ++G) {
        const int n_lo = nw + 4 * G + hi;
        ABTile wa = load_w_tile4(W, j, n_lo, n_lo + 2, bl);
        if (MODE == 0) {
          acc = __builtin_amdgcn_wmma_f32_16x16x32_bf16(
              false, wa.bf, false, bx4[G].bf, (short)0, acc, false, false);
        } else {
          // Fold c[b,j,n] into the x operand, then one WMMA covers 4 n.
          const float c_lo = ldsL[((w * 8 + 4 * G + hi) * J_ + j) * 16 + bl];
          const float c_hi = ldsL[((w * 8 + 4 * G + 2 + hi) * J_ + j) * 16 + bl];
          ABTile bcx;
          #pragma unroll
          for (int k = 0; k < 8; ++k) {
            bcx.us[k]     = f2bf(c_lo * xf[G][k]);
            bcx.us[8 + k] = f2bf(c_hi * xf[G][8 + k]);
          }
          acc = __builtin_amdgcn_wmma_f32_16x16x32_bf16(
              false, wa.bf, false, bcx.bf, (short)0, acc, false, false);
        }
      }
      if (MODE == 0) {
        #pragma unroll
        for (int r = 0; r < 8; ++r) acc[r] *= (1.0f / 32.0f);
      }
      #pragma unroll
      for (int r = 0; r < 8; ++r)
        ldsS[(w * 8 + jg) * 256 + (r + 8 * hi) * 16 + bl] = acc[r];
    }
    __syncthreads();
    // Cross-wave reduce this j-group and emit deterministic per-chunk partials.
    for (int idx = tid; idx < 2048; idx += 128) {
      float ssum = ldsS[idx] + ldsS[2048 + idx] + ldsS[4096 + idx] + ldsS[6144 + idx];
      spart[(size_t)p_idx * 8192 + g * 2048 + idx] = ssum;
    }
    __syncthreads();
  }
}

// Reduce the 64 n-chunk partials per (b,j), squash over d, write v.
__global__ void __launch_bounds__(256)
caps_squash(const float* __restrict__ spart, float* __restrict__ vout) {
  const int t    = threadIdx.x;
  const int pair = blockIdx.x * 16 + (t >> 4);  // (b,j) pair: 0..2047
  const int d    = t & 15;
  const int b    = pair >> 5;
  const int j    = pair & 31;
  const int bt   = b >> 4;
  const int bl   = b & 15;
  const float* sp = spart + (size_t)bt * 64 * 8192 + j * 256 + d * 16 + bl;
  float s = 0.0f;
  #pragma unroll 4
  for (int c = 0; c < 64; ++c) s += sp[(size_t)c * 8192];
  float ssq = s * s;
  ssq += __shfl_xor(ssq, 1, 32);
  ssq += __shfl_xor(ssq, 2, 32);
  ssq += __shfl_xor(ssq, 4, 32);
  ssq += __shfl_xor(ssq, 8, 32);   // sum over the 16 d-lanes of this pair
  vout[(size_t)b * (J_ * D_) + j * D_ + d] = s * rsqrtf(ssq + 1e-7f);
}

extern "C" void kernel_launch(void* const* d_in, const int* in_sizes, int n_in,
                              void* d_out, int out_size, void* d_ws, size_t ws_size,
                              hipStream_t stream) {
  (void)in_sizes; (void)n_in; (void)out_size; (void)ws_size;
  const float* x = (const float*)d_in[0];
  const float* W = (const float*)d_in[1];
  float* vout  = (float*)d_out;
  float* spart = (float*)d_ws;                                     // 8 MiB
  float* Lws   = (float*)((char*)d_ws + (size_t)8 * 1024 * 1024);  // 16 MiB

  const dim3 rgrid(64, 4);          // 64 n-chunks x 4 b-tiles = 256 WGs
  const int  smem = (8192 + 16384) * (int)sizeof(float);  // 96 KiB dynamic LDS

  // Routing iteration 0 (uniform coupling), 1, 2 — each followed by squash.
  caps_route<0><<<rgrid, 128, smem, stream>>>(x, W, vout, spart, Lws);
  caps_squash<<<128, 256, 0, stream>>>(spart, vout);
  caps_route<1><<<rgrid, 128, smem, stream>>>(x, W, vout, spart, Lws);
  caps_squash<<<128, 256, 0, stream>>>(spart, vout);
  caps_route<2><<<rgrid, 128, smem, stream>>>(x, W, vout, spart, Lws);
  caps_squash<<<128, 256, 0, stream>>>(spart, vout);
}
